// GatedDeltaNet_27247272525829
// MI455X (gfx1250) — matrix operand order repaired and checked by reading, not compiled, not run on hardware
//
#include <hip/hip_runtime.h>
#include <cstdint>
#include <cstddef>

// ---- problem constants (match reference) ----
#define B_   2
#define T_   2048
#define D_   2048
#define NK_  16
#define NV_  32
#define KD_  128
#define VD_  128
#define CH_  8192            // 2*NK*KD + NV*VD
#define BT_  (B_*T_)
#define ROT_ 32
#define EPS_ 1e-6f
#define REP_ (NV_/NK_)       // 2

typedef __attribute__((ext_vector_type(16))) __bf16 v16bf;
typedef __attribute__((ext_vector_type(8)))  __bf16 v8bf;
typedef __attribute__((ext_vector_type(8)))  float  v8f;

__device__ __forceinline__ float sigmoidf_(float x) { return 1.f / (1.f + __expf(-x)); }

// ---------------------------------------------------------------------------
// CDNA5 async global->LDS copy (ASYNCcnt-tracked, no VGPR data round trip).
// LDS byte address = low 32 bits of the flat pointer (LDS aperture truncation).
// ---------------------------------------------------------------------------
__device__ __forceinline__ void async_copy_b128(const void* gaddr, void* lds) {
  unsigned long long ga = (unsigned long long)gaddr;
  unsigned la = (unsigned)(uintptr_t)lds;
  asm volatile("global_load_async_to_lds_b128 %0, %1, off" :: "v"(la), "v"(ga) : "memory");
}
__device__ __forceinline__ void wait_async0() {
  asm volatile("s_wait_asynccnt 0x0" ::: "memory");
}

// ---------------------------------------------------------------------------
// Tensor Data Mover path (6-arg builtin on this toolchain). Loads `nrows`
// f32 elements with stride `stride_elems` (in elements) from global into
// contiguous LDS at lds_off.  D# per CDNA5 ISA ch. 8 (2D tile: 1 x nrows).
// ---------------------------------------------------------------------------
#if __has_builtin(__builtin_amdgcn_tensor_load_to_lds)
#define HAS_TDM_ 1
typedef __attribute__((ext_vector_type(4))) unsigned int u32x4;
typedef __attribute__((ext_vector_type(8))) int i32x8;
typedef __attribute__((ext_vector_type(4))) int i32x4;

__device__ __forceinline__ void tdm_load_strided_f32(const void* gaddr, unsigned lds_off,
                                                     unsigned nrows, unsigned stride_elems) {
  unsigned long long ga = (unsigned long long)gaddr;
  u32x4 g0;
  g0[0] = 1u;                                                // count=1 (valid user D#)
  g0[1] = lds_off;                                           // lds_addr (bytes)
  g0[2] = (unsigned)(ga & 0xffffffffu);                      // global_addr[31:0]
  g0[3] = (unsigned)((ga >> 32) & 0x01ffffffu) | (2u << 30); // global_addr[56:32] | type=2
  i32x8 g1;
  g1[0] = (int)(2u << 16);         // wg_mask=0, data_size=2 (4 bytes)
  g1[1] = (int)(1u << 16);         // tensor_dim0 lo16 = 1
  g1[2] = (int)(nrows << 16);      // tensor_dim0 hi = 0, tensor_dim1 lo16 = nrows
  g1[3] = (int)(1u << 16);         // tensor_dim1 hi = 0, tile_dim0 = 1
  g1[4] = (int)nrows;              // tile_dim1 = nrows, tile_dim2 = 0
  g1[5] = (int)stride_elems;       // tensor_dim0_stride[31:0]
  g1[6] = 0;                       // stride hi / dim1_stride lo
  g1[7] = 0;
  i32x4 z4 = {0, 0, 0, 0};
  i32x8 z8 = {0, 0, 0, 0, 0, 0, 0, 0};
  __builtin_amdgcn_tensor_load_to_lds(g0, g1, z4, z4, z8, 0);
}
#else
#define HAS_TDM_ 0
#endif

// ---------------------------------------------------------------------------
// tm[bt] = any(attn_mask[bt, :])
// ---------------------------------------------------------------------------
__global__ void tm_kernel(const unsigned char* __restrict__ mask, float* __restrict__ tm) {
  int bt = blockIdx.x;
  __shared__ int anyv;
  if (threadIdx.x == 0) anyv = 0;
  __syncthreads();
  int found = 0;
  const unsigned char* row = mask + (long)bt * T_;
  for (int j = threadIdx.x; j < T_; j += blockDim.x) found |= (row[j] != 0);
  if (found) atomicOr(&anyv, 1);
  __syncthreads();
  if (threadIdx.x == 0) tm[bt] = anyv ? 1.f : 0.f;
}

// ---------------------------------------------------------------------------
// Xb[bt,d] = bf16(x[bt,d] * tm[bt])
// ---------------------------------------------------------------------------
__global__ void convert_x_kernel(const float* __restrict__ x, const float* __restrict__ tm,
                                 __bf16* __restrict__ Xb) {
  long i = (long)blockIdx.x * blockDim.x + threadIdx.x;
  if (i < (long)BT_ * D_) {
    long bt = i / D_;
    Xb[i] = (__bf16)(x[i] * tm[bt]);
  }
}

// ---------------------------------------------------------------------------
// Wt[n,k] = bf16(W[k,n])   (W is [Kdim][Ndim] row-major)
// ---------------------------------------------------------------------------
__global__ void transpose_convert_kernel(const float* __restrict__ W, __bf16* __restrict__ Wt,
                                         int Kdim, int Ndim) {
  long i = (long)blockIdx.x * blockDim.x + threadIdx.x;
  if (i < (long)Kdim * Ndim) {
    long n = i / Kdim;
    long k = i - n * Kdim;
    Wt[i] = (__bf16)W[k * (long)Ndim + n];
  }
}

// ---------------------------------------------------------------------------
// Tiled bf16 WMMA GEMM: C[M][N] = A[M][K] * Bt[N][K]^T   (f32 accumulate)
// 256 threads (8 waves), tile 128x128, K-chunk 32, DOUBLE-BUFFERED LDS with
// async global->LDS DMA overlapped against the 8 WMMAs of the current chunk.
// ---------------------------------------------------------------------------
#define GTM 128
#define GTN 128
#define GTK 32

__global__ __launch_bounds__(256) void wmma_gemm_bf16(
    const __bf16* __restrict__ A, const __bf16* __restrict__ Bt,
    float* __restrict__ C, int M, int N, int K) {
  __shared__ __bf16 As[2][GTM * GTK];
  __shared__ __bf16 Bs[2][GTN * GTK];
  const int tid  = threadIdx.x;
  const int lane = tid & 31;
  const int wave = tid >> 5;
  const int wm = (wave & 1) * 64;
  const int wn = (wave >> 1) * 32;
  const long m0 = (long)blockIdx.y * GTM;
  const long n0 = (long)blockIdx.x * GTN;

  v8f acc[4][2];
  const v8f z8 = {0.f, 0.f, 0.f, 0.f, 0.f, 0.f, 0.f, 0.f};
#pragma unroll
  for (int i = 0; i < 4; ++i)
#pragma unroll
    for (int j = 0; j < 2; ++j) acc[i][j] = z8;

  const int frow = lane & 15;            // fragment row within 16
  const int akb  = (lane >> 4) * 8;      // A K-subchunk base (ISA 16-bit A layout)
  const int bkb  = (lane >> 4) * 8;      // B K-subchunk base (same layout as A)
  const int er   = (tid * 8) >> 5;       // staging row (0..63), +64 for rep 1
  const int ec   = (tid * 8) & 31;       // staging col

  // issue one tile (A+B) of async DMA into LDS buffer `sel`
  auto stage = [&](int k0, int sel) {
#pragma unroll
    for (int rep = 0; rep < 2; ++rep) {
      int r = er + rep * 64;
      int e = tid * 8 + rep * 2048;
      async_copy_b128(&A[(m0 + r) * (long)K + k0 + ec], &As[sel][e]);
      async_copy_b128(&Bt[(n0 + r) * (long)K + k0 + ec], &Bs[sel][e]);
    }
  };

  stage(0, 0);
  wait_async0();
  __syncthreads();

  const int nch = K / GTK;
  for (int ci = 0; ci < nch; ++ci) {
    const int cur = ci & 1;
    if (ci + 1 < nch) stage((ci + 1) * GTK, cur ^ 1);  // DMA next chunk (overlapped)

    const __bf16* Ab = &As[cur][0];
    const __bf16* Bb = &Bs[cur][0];
    v16bf afrag[4], bfrag[2];
#pragma unroll
    for (int mt = 0; mt < 4; ++mt) {
      int row = wm + mt * 16 + frow;
      v8bf lo = *(const v8bf*)&Ab[row * GTK + akb];
      v8bf hi = *(const v8bf*)&Ab[row * GTK + akb + 16];
      afrag[mt] = __builtin_shufflevector(lo, hi, 0,1,2,3,4,5,6,7,8,9,10,11,12,13,14,15);
    }
#pragma unroll
    for (int nt = 0; nt < 2; ++nt) {
      int rown = wn + nt * 16 + frow;
      v8bf lo = *(const v8bf*)&Bb[rown * GTK + bkb];
      v8bf hi = *(const v8bf*)&Bb[rown * GTK + bkb + 16];
      bfrag[nt] = __builtin_shufflevector(lo, hi, 0,1,2,3,4,5,6,7,8,9,10,11,12,13,14,15);
    }
#pragma unroll
    for (int mt = 0; mt < 4; ++mt)
#pragma unroll
      for (int nt = 0; nt < 2; ++nt)
        acc[mt][nt] = __builtin_amdgcn_wmma_f32_16x16x32_bf16(
            false, afrag[mt], false, bfrag[nt], (short)0, acc[mt][nt], false, false);

    wait_async0();        // next buffer fully in LDS (own wave's DMA)
    __syncthreads();      // all waves done reading `cur`, all DMAs visible
  }

  // C/D layout: VGPR r, lanes 0-15 -> M=r, lanes 16-31 -> M=8+r, N=lane%16
  const int roff = (lane >> 4) * 8;
  const int ccol = lane & 15;
#pragma unroll
  for (int mt = 0; mt < 4; ++mt)
#pragma unroll
    for (int nt = 0; nt < 2; ++nt)
#pragma unroll
      for (int r = 0; r < 8; ++r) {
        long row = m0 + wm + mt * 16 + roff + r;
        long col = n0 + wn + nt * 16 + ccol;
        C[row * (long)N + col] = acc[mt][nt][r];
      }
}

// ---------------------------------------------------------------------------
// In-place causal depthwise conv (width 4) over time + SiLU.
// ---------------------------------------------------------------------------
__global__ void conv_silu_kernel(float* __restrict__ qkv, const float* __restrict__ ck) {
  int idx = blockIdx.x * blockDim.x + threadIdx.x;
  if (idx >= B_ * CH_) return;
  int b = idx / CH_;
  int c = idx - b * CH_;
  float k0 = ck[0 * CH_ + c], k1 = ck[1 * CH_ + c], k2 = ck[2 * CH_ + c], k3 = ck[3 * CH_ + c];
  float h1 = 0.f, h2 = 0.f, h3 = 0.f;
  float* base = qkv + (long)b * T_ * CH_ + c;
  for (int t = 0; t < T_; ++t) {
    float cur = base[(long)t * CH_];
    float y = k0 * cur + k1 * h1 + k2 * h2 + k3 * h3;
    h3 = h2; h2 = h1; h1 = cur;
    base[(long)t * CH_] = y * sigmoidf_(y);
  }
}

// ---------------------------------------------------------------------------
// beta = sigmoid(x@W_b)*tm ;  g = tm>0 ? -exp(A_log)*softplus(x@W_a + dt_bias) : 0
// ---------------------------------------------------------------------------
__global__ void beta_g_kernel(const float* __restrict__ x, const float* __restrict__ tm,
                              const float* __restrict__ Wb, const float* __restrict__ Wa,
                              const float* __restrict__ A_log, const float* __restrict__ dt_bias,
                              float* __restrict__ beta, float* __restrict__ g) {
  int bt = blockIdx.x;
  int j = threadIdx.x;
  float t = tm[bt];
  const float* xr = x + (long)bt * D_;
  if (j < NV_) {
    float s = 0.f;
    for (int i = 0; i < D_; ++i) s += xr[i] * Wb[(long)i * NV_ + j];
    s *= t;
    beta[(long)bt * NV_ + j] = sigmoidf_(s) * t;
  } else {
    int h = j - NV_;
    float s = 0.f;
    for (int i = 0; i < D_; ++i) s += xr[i] * Wa[(long)i * NV_ + h];
    s *= t;
    float aa = s + dt_bias[h];
    float sp = (aa > 20.f) ? aa : log1pf(__expf(aa));
    float gg = -__expf(A_log[h]) * sp;
    g[(long)bt * NV_ + h] = (t > 0.f) ? gg : 0.f;
  }
}

// ---------------------------------------------------------------------------
// RoPE (first 32 dims) + tm mask + L2 normalize (q additionally * KD^-0.5)
// ---------------------------------------------------------------------------
__global__ void rope_norm_kernel(const float* __restrict__ qkv, const int* __restrict__ positions,
                                 const float* __restrict__ tm,
                                 float* __restrict__ qn, float* __restrict__ kn) {
  int idx = blockIdx.x;
  int which = idx & 1;                 // 0 = q, 1 = k
  int bth = idx >> 1;
  int hk = bth % NK_;
  int bt = bth / NK_;
  int d = threadIdx.x;                 // 0..127
  const float* src = qkv + (long)bt * CH_ + (long)which * NK_ * KD_ + (long)hk * KD_;
  float val = src[d] * tm[bt];

  __shared__ float vec[KD_];
  __shared__ float red[KD_];
  vec[d] = val;
  __syncthreads();

  float out;
  const float pos = (float)positions[bt];
  const float cexp = 1.0073809757f;    // log(1e7)/16
  if (d < ROT_ / 2) {
    float inv = __expf(-(float)d * cexp);
    float ang = pos * inv;
    out = vec[d] * __cosf(ang) - vec[d + ROT_ / 2] * __sinf(ang);
  } else if (d < ROT_) {
    int i = d - ROT_ / 2;
    float inv = __expf(-(float)i * cexp);
    float ang = pos * inv;
    out = vec[d] * __cosf(ang) + vec[d - ROT_ / 2] * __sinf(ang);
  } else {
    out = val;
  }

  red[d] = out * out;
  __syncthreads();
  for (int s = 64; s > 0; s >>= 1) {
    if (d < s) red[d] += red[d + s];
    __syncthreads();
  }
  float nrm = fmaxf(sqrtf(red[0]), EPS_);
  float res = out / nrm;
  if (which == 0) res *= 0.08838834764831845f;  // KD^-0.5
  float* dst = (which == 0) ? qn : kn;
  dst[((long)bt * NK_ + hk) * KD_ + d] = res;
}

// ---------------------------------------------------------------------------
// Sequential delta-rule scan. One block per (b, h). 128x128 f32 state in LDS.
// Per-head g/beta gate vectors DMA'd into LDS up front by the TDM.
// ---------------------------------------------------------------------------
__global__ __launch_bounds__(256) void scan_kernel(
    const float* __restrict__ qn, const float* __restrict__ kn,
    const float* __restrict__ qkv, const float* __restrict__ tm,
    const float* __restrict__ g, const float* __restrict__ beta,
    float* __restrict__ o) {
  extern __shared__ float smem[];
  float* S       = smem;                    // KD_*VD_
  float* kvec    = S + KD_ * VD_;           // KD_
  float* qvec    = kvec + KD_;              // KD_
  float* vvec    = qvec + KD_;              // VD_
  float* kv_part = vvec + VD_;              // 2*VD_
  float* o_part  = kv_part + 2 * VD_;       // 2*VD_
  float* gh      = o_part + 2 * VD_;        // T_
  float* bh      = gh + T_;                 // T_

  int bhid = blockIdx.x;
  int b = bhid / NV_;
  int h = bhid - b * NV_;
  int kh = h / REP_;
  int tid = threadIdx.x;
  int v = tid & 127;
  int half = tid >> 7;
  int ks = half * 64;

  for (int i = tid; i < KD_ * VD_; i += 256) S[i] = 0.f;

#if HAS_TDM_
  if (tid < 32) {  // one wave issues the tensor DMA; TENSORcnt is per-wave
    tdm_load_strided_f32(&g[(long)b * T_ * NV_ + h], (unsigned)(uintptr_t)gh, T_, NV_);
    tdm_load_strided_f32(&beta[(long)b * T_ * NV_ + h], (unsigned)(uintptr_t)bh, T_, NV_);
    __builtin_amdgcn_s_wait_tensorcnt(0);
  }
#else
  for (int t = tid; t < T_; t += 256) {
    gh[t] = g[((long)b * T_ + t) * NV_ + h];
    bh[t] = beta[((long)b * T_ + t) * NV_ + h];
  }
#endif
  __syncthreads();

  for (int t = 0; t < T_; ++t) {
    long bt = (long)b * T_ + t;
    if (tid < KD_) {
      kvec[tid] = kn[(bt * NK_ + kh) * KD_ + tid];
      qvec[tid] = qn[(bt * NK_ + kh) * KD_ + tid];
    } else {
      int vv = tid - KD_;
      vvec[vv] = qkv[bt * CH_ + 2 * NK_ * KD_ + (long)h * VD_ + vv] * tm[bt];
    }
    __syncthreads();

    float eg = __expf(gh[t]);
    float bt_ = bh[t];

    float p = 0.f;
#pragma unroll 8
    for (int k = ks; k < ks + 64; ++k) p += S[k * VD_ + v] * kvec[k];
    kv_part[half * VD_ + v] = p * eg;
    __syncthreads();

    float kvv = kv_part[v] + kv_part[VD_ + v];
    float delta = bt_ * (vvec[v] - kvv);

    float po = 0.f;
#pragma unroll 8
    for (int k = ks; k < ks + 64; ++k) {
      float s = S[k * VD_ + v] * eg + kvec[k] * delta;
      S[k * VD_ + v] = s;
      po += s * qvec[k];
    }
    o_part[half * VD_ + v] = po;
    __syncthreads();

    if (half == 0) o[(bt * NV_ + h) * VD_ + v] = o_part[v] + o_part[VD_ + v];
    __syncthreads();
  }
}

// ---------------------------------------------------------------------------
// gated = o * silu(z*tm); RMS-norm over VD; write bf16 normed[bt, h*VD+v]
// ---------------------------------------------------------------------------
__global__ void gated_norm_kernel(const float* __restrict__ o, const float* __restrict__ Z,
                                  const float* __restrict__ tm,
                                  const float* __restrict__ norm_scale,
                                  __bf16* __restrict__ normed) {
  int bth = blockIdx.x;
  int h = bth % NV_;
  long bt = bth / NV_;
  int v = threadIdx.x;
  float zz = Z[bt * (NV_ * VD_) + (long)h * VD_ + v] * tm[bt];
  float gated = o[(bt * NV_ + h) * VD_ + v] * (zz * sigmoidf_(zz));
  __shared__ float red[VD_];
  red[v] = gated * gated;
  __syncthreads();
  for (int s = 64; s > 0; s >>= 1) {
    if (v < s) red[v] += red[v + s];
    __syncthreads();
  }
  float var = red[0] * (1.f / VD_);
  float nm = gated * rsqrtf(var + EPS_) * (1.f + norm_scale[v]);
  normed[bt * (NV_ * VD_) + (long)h * VD_ + v] = (__bf16)nm;
}

// ---------------------------------------------------------------------------
extern "C" void kernel_launch(void* const* d_in, const int* in_sizes, int n_in,
                              void* d_out, int out_size, void* d_ws, size_t ws_size,
                              hipStream_t stream) {
  const float*         x         = (const float*)d_in[0];
  const int*           positions = (const int*)d_in[1];
  const unsigned char* attn_mask = (const unsigned char*)d_in[2];
  const float*         W_qkv     = (const float*)d_in[3];
  const float*         W_z       = (const float*)d_in[4];
  const float*         W_b       = (const float*)d_in[5];
  const float*         W_a       = (const float*)d_in[6];
  const float*         convk     = (const float*)d_in[7];
  const float*         A_log     = (const float*)d_in[8];
  const float*         dt_bias   = (const float*)d_in[9];
  const float*         nscale    = (const float*)d_in[10];
  const float*         out_W     = (const float*)d_in[11];
  float* out = (float*)d_out;
  uint8_t* ws = (uint8_t*)d_ws;

  size_t off = 0;
  auto take = [&](size_t bytes) -> size_t {
    size_t o = off;
    off = (off + bytes + 255) & ~(size_t)255;
    return o;
  };
  float*  tm      = (float*)(ws + take((size_t)BT_ * 4));
  __bf16* Xb      = (__bf16*)(ws + take((size_t)BT_ * D_ * 2));
  __bf16* Wqkv_t  = (__bf16*)(ws + take((size_t)CH_ * D_ * 2));
  __bf16* Wz_t    = (__bf16*)(ws + take((size_t)(NV_ * VD_) * D_ * 2));
  __bf16* Wout_t  = (__bf16*)(ws + take((size_t)D_ * (NV_ * VD_) * 2));
  float*  QKV     = (float*)(ws + take((size_t)BT_ * CH_ * 4));
  float*  Zbuf    = (float*)(ws + take((size_t)BT_ * NV_ * VD_ * 4));
  float*  betab   = (float*)(ws + take((size_t)BT_ * NV_ * 4));
  float*  gbuf    = (float*)(ws + take((size_t)BT_ * NV_ * 4));
  float*  qn      = (float*)(ws + take((size_t)BT_ * NK_ * KD_ * 4));
  float*  kn      = (float*)(ws + take((size_t)BT_ * NK_ * KD_ * 4));
  float*  obuf    = (float*)(ws + take((size_t)BT_ * NV_ * VD_ * 4));
  __bf16* normed  = (__bf16*)(ws + take((size_t)BT_ * NV_ * VD_ * 2));
  (void)ws_size; (void)in_sizes; (void)n_in; (void)out_size;

  // 1. mask reduction + input conversion
  tm_kernel<<<BT_, 256, 0, stream>>>(attn_mask, tm);
  convert_x_kernel<<<(BT_ * D_) / 256, 256, 0, stream>>>(x, tm, Xb);

  // 2. weight transpose + bf16 convert
  transpose_convert_kernel<<<((long)D_ * CH_) / 256, 256, 0, stream>>>(W_qkv, Wqkv_t, D_, CH_);
  transpose_convert_kernel<<<((long)D_ * NV_ * VD_) / 256, 256, 0, stream>>>(W_z, Wz_t, D_, NV_ * VD_);
  transpose_convert_kernel<<<((long)NV_ * VD_ * D_) / 256, 256, 0, stream>>>(out_W, Wout_t, NV_ * VD_, D_);

  // 3. QKV = Xb @ W_qkv   (M=4096, N=8192, K=2048)
  wmma_gemm_bf16<<<dim3(CH_ / GTN, BT_ / GTM), 256, 0, stream>>>(Xb, Wqkv_t, QKV, BT_, CH_, D_);

  // 4. depthwise causal conv + SiLU (in place)
  conv_silu_kernel<<<(B_ * CH_) / 256, 256, 0, stream>>>(QKV, convk);

  // 5. Z = Xb @ W_z   (M=4096, N=4096, K=2048)
  wmma_gemm_bf16<<<dim3((NV_ * VD_) / GTN, BT_ / GTM), 256, 0, stream>>>(Xb, Wz_t, Zbuf, BT_, NV_ * VD_, D_);

  // 6. beta / g gates
  beta_g_kernel<<<BT_, 64, 0, stream>>>(x, tm, W_b, W_a, A_log, dt_bias, betab, gbuf);

  // 7. RoPE + L2 norm for q and k
  rope_norm_kernel<<<BT_ * NK_ * 2, 128, 0, stream>>>(QKV, positions, tm, qn, kn);

  // 8. sequential delta-rule scan (LDS-resident state per head, TDM gate preload)
  size_t scan_lds = (size_t)(KD_ * VD_ + 2 * KD_ + VD_ + 4 * VD_ + 2 * T_) * sizeof(float);
  scan_kernel<<<B_ * NV_, 256, scan_lds, stream>>>(qn, kn, QKV, tm, gbuf, betab, obuf);

  // 9. gate + RMS norm -> bf16
  gated_norm_kernel<<<BT_ * NV_, 128, 0, stream>>>(obuf, Zbuf, tm, nscale, normed);

  // 10. output projection: out = normed @ out_W   (M=4096, N=2048, K=4096)
  wmma_gemm_bf16<<<dim3(D_ / GTN, BT_ / GTM), 256, 0, stream>>>(normed, Wout_t, out, BT_, D_, NV_ * VD_);
}
